// BiGCN_8847632630379
// MI455X (gfx1250) — matrix operand order repaired and checked by reading, not compiled, not run on hardware
//
#include <hip/hip_runtime.h>
#include <hip/hip_bf16.h>

// ---------------------------------------------------------------------------
// BiGCN on gfx1250: big GEMMs via v_wmma_f32_16x16x32_f16 (register double-
// buffered, 32x64 tile per wave), graph aggregation via f32 atomics after the
// GEMM (aggregate over 256 features), exact-f32 side paths for norms / root
// projections / readout.
// ---------------------------------------------------------------------------

typedef __attribute__((ext_vector_type(16))) _Float16 v16h;
typedef __attribute__((ext_vector_type(8)))  _Float16 h8;
typedef __attribute__((ext_vector_type(4)))  _Float16 h4;
typedef __attribute__((ext_vector_type(8)))  float    v8f;

#define NPG   128      // nodes per graph
#define IN_F  768
#define HF    256

// ---------------------------------------------------------------------------
// Degree accumulation: deg[dst[e]] += 1
// ---------------------------------------------------------------------------
__global__ void bigcn_deg_kernel(const int* __restrict__ dst, int E,
                                 float* __restrict__ deg) {
  int e = blockIdx.x * blockDim.x + threadIdx.x;
  if (e < E) atomicAdd(&deg[dst[e]], 1.0f);
}

// deg -> norm = rsqrt(deg), in place
__global__ void bigcn_norm_kernel(float* __restrict__ d, int n) {
  int i = blockIdx.x * blockDim.x + threadIdx.x;
  if (i < n) d[i] = rsqrtf(d[i]);
}

// ---------------------------------------------------------------------------
// f32 -> f16 conversion, 4 elements / thread
// ---------------------------------------------------------------------------
__global__ void bigcn_cvt_f16_kernel(const float* __restrict__ x,
                                     _Float16* __restrict__ y, int n4) {
  int i = blockIdx.x * blockDim.x + threadIdx.x;
  if (i < n4) {
    float4 v = ((const float4*)x)[i];
    h4 o;
    o[0] = (_Float16)v.x; o[1] = (_Float16)v.y;
    o[2] = (_Float16)v.z; o[3] = (_Float16)v.w;
    ((h4*)y)[i] = o;
  }
}

// ---------------------------------------------------------------------------
// Pack a [Krows x 256] f32 weight (row-major, rows = K) into the WMMA
// B-fragment f16 layout this GEMM loads (contiguous k order): Wp[(k/16)*256 + n][16] with t = k%16.
// Each lane's B fragment is then 32 contiguous bytes.
// ---------------------------------------------------------------------------
__global__ void bigcn_pack_w_kernel(const float* __restrict__ W, int Krows,
                                    int rowOff, _Float16* __restrict__ Wp) {
  int idx = blockIdx.x * blockDim.x + threadIdx.x;
  if (idx < Krows * HF) {
    int k = idx >> 8;           // row (K index)
    int n = idx & 255;          // column
    Wp[((((size_t)(k >> 4)) << 8) + n) * 16 + (k & 15)] =
        (_Float16)W[(size_t)(rowOff + k) * HF + n];
  }
}

// ---------------------------------------------------------------------------
// WMMA GEMM: C[M x 256] = A[M x K] (f16, row-major) @ Wp (packed f16).
// Block = 128 threads = 4 waves; wave w owns columns [64w, 64w+64).
// Each wave computes a 32x64 C tile: 2 M-tiles x 4 N-tiles = 8 accumulators.
// Register double-buffering: next k-block's fragments load while the current
// k-block's 8 WMMAs issue.  gridDim.x = M / 32; K % 64 == 0 required.
// ---------------------------------------------------------------------------
struct BigcnFrags {
  h8 alo[2], ahi[2];   // 2 M-tiles
  h8 blo[4], bhi[4];   // 4 N-tiles
};

__global__ __launch_bounds__(128)
void bigcn_gemm_wmma_kernel(const _Float16* __restrict__ A, int K,
                            const _Float16* __restrict__ Wp,
                            float* __restrict__ C) {
  const int lane = threadIdx.x & 31;
  const int wave = threadIdx.x >> 5;
  const int m0   = blockIdx.x << 5;       // 32 rows per block
  const int n0   = wave << 6;             // 64 cols per wave

  const int koff = (lane & 16) ? 16 : 0;  // A: K sub-offset per lane group
  const int kbo  = (lane & 16) ? 1 : 0;   // B: 16-wide K sub-block per group
  const int ncol = lane & 15;

  const _Float16* arow0 = A + (size_t)(m0 + (lane & 15)) * K;
  const _Float16* arow1 = arow0 + (size_t)16 * K;
  const _Float16* wbase = Wp + ((size_t)(n0 + ncol)) * 16;

  v8f acc[2][4];
#pragma unroll
  for (int m = 0; m < 2; ++m)
#pragma unroll
    for (int j = 0; j < 4; ++j) acc[m][j] = v8f{};

  // ---- fragment load: A in contiguous K order (lanes 0-15 hold
  // K 0-15, lanes 16-31 hold K 16-31); B frag is 32 contiguous
  // bytes per lane in the packed weight layout.
  auto load_frags = [&](BigcnFrags& f, int k0) {
    f.alo[0] = *(const h8*)(arow0 + k0 + koff);
    f.ahi[0] = *(const h8*)(arow0 + k0 + 8 + koff);
    f.alo[1] = *(const h8*)(arow1 + k0 + koff);
    f.ahi[1] = *(const h8*)(arow1 + k0 + 8 + koff);
    const _Float16* wb = wbase + (((size_t)((k0 >> 4) + kbo)) << 12);
#pragma unroll
    for (int j = 0; j < 4; ++j) {
      f.blo[j] = *(const h8*)(wb + (j << 8));
      f.bhi[j] = *(const h8*)(wb + (j << 8) + 8);
    }
  };

  auto compute = [&](const BigcnFrags& f) {
#pragma unroll
    for (int m = 0; m < 2; ++m) {
      v16h a;
#pragma unroll
      for (int i = 0; i < 8; ++i) { a[i] = f.alo[m][i]; a[i + 8] = f.ahi[m][i]; }
#pragma unroll
      for (int j = 0; j < 4; ++j) {
        v16h b;
#pragma unroll
        for (int i = 0; i < 8; ++i) { b[i] = f.blo[j][i]; b[i + 8] = f.bhi[j][i]; }
        acc[m][j] = __builtin_amdgcn_wmma_f32_16x16x32_f16(
            false, a, false, b, (short)0, acc[m][j], false, false);
      }
    }
  };

  BigcnFrags f0, f1;
  load_frags(f0, 0);
  for (int k0 = 0; k0 < K; k0 += 64) {
    load_frags(f1, k0 + 32);        // overlap with compute(f0)
    compute(f0);
    if (k0 + 64 < K) load_frags(f0, k0 + 64);
    compute(f1);
  }

  // C/D layout: VGPR r -> (M = r + (lane<16 ? 0 : 8), N = lane & 15)
  const int mb = m0 + ((lane & 16) ? 8 : 0);
#pragma unroll
  for (int m = 0; m < 2; ++m)
#pragma unroll
    for (int j = 0; j < 4; ++j) {
      const int nc = n0 + (j << 4) + ncol;
#pragma unroll
      for (int r = 0; r < 8; ++r)
        C[(size_t)(mb + (m << 4) + r) * HF + nc] = acc[m][j][r];
    }
}

// ---------------------------------------------------------------------------
// Edge scatter: agg[d, c] += norm[s] * (val[s, c] + rG[g(s), c])
// grid = E blocks, 256 threads (one per feature column).
// ---------------------------------------------------------------------------
__global__ void bigcn_scatter_kernel(const int* __restrict__ src,
                                     const int* __restrict__ dst,
                                     const float* __restrict__ val,
                                     const float* __restrict__ norm,
                                     const float* __restrict__ rG,
                                     float* __restrict__ agg) {
  int e = blockIdx.x;
  int c = threadIdx.x;
  int s = src[e], d = dst[e];
  float v = val[(size_t)s * HF + c];
  if (rG) v += rG[(size_t)((s / NPG)) * HF + c];
  atomicAdd(&agg[(size_t)d * HF + c], norm[s] * v);
}

// ---------------------------------------------------------------------------
// Layer-1 epilogue: h = relu(norm * agg + b1) -> f16; root rows kept in f32.
// grid = N blocks, 256 threads.
// ---------------------------------------------------------------------------
__global__ void bigcn_finish_h_kernel(const float* __restrict__ agg,
                                      const float* __restrict__ norm,
                                      const float* __restrict__ b1,
                                      _Float16* __restrict__ h16,
                                      float* __restrict__ rootH) {
  int i = blockIdx.x;
  int c = threadIdx.x;
  float v = fmaxf(norm[i] * agg[(size_t)i * HF + c] + b1[c], 0.0f);
  h16[(size_t)i * HF + c] = (_Float16)v;
  if ((i & (NPG - 1)) == 0)
    rootH[(size_t)(i / NPG) * HF + c] = v;
}

// ---------------------------------------------------------------------------
// rG[g, n] = sum_k X[root_g, k] * W2[256 + k, n]   (exact f32, tiny GEMM)
// grid = G blocks, 256 threads.
// ---------------------------------------------------------------------------
__global__ void bigcn_rootproj_kernel(const float* __restrict__ X,
                                      const float* __restrict__ W2,
                                      float* __restrict__ rG) {
  __shared__ float xs[256];
  int g = blockIdx.x, n = threadIdx.x;
  const float* xr = X + (size_t)g * NPG * IN_F;  // root = first node of graph
  float acc = 0.0f;
  for (int kb = 0; kb < IN_F; kb += 256) {
    xs[n] = xr[kb + n];
    __syncthreads();
#pragma unroll 8
    for (int t = 0; t < 256; ++t)
      acc += xs[t] * W2[(size_t)(HF + kb + t) * HF + n];
    __syncthreads();
  }
  rG[(size_t)g * HF + n] = acc;
}

// ---------------------------------------------------------------------------
// Readout: out[g, colbase + c]       = mean_j relu(norm*agg2 + b2)
//          out[g, colbase + 256 + c] = rootH[g, c]
// grid = G blocks, 256 threads.
// ---------------------------------------------------------------------------
__global__ void bigcn_readout_kernel(const float* __restrict__ agg,
                                     const float* __restrict__ norm,
                                     const float* __restrict__ b2,
                                     const float* __restrict__ rootH,
                                     float* __restrict__ out, int colbase) {
  int g = blockIdx.x, c = threadIdx.x;
  float bc = b2[c];
  float s = 0.0f;
  const float* ag = agg + (size_t)g * NPG * HF;
  const float* nm = norm + (size_t)g * NPG;
#pragma unroll 4
  for (int j = 0; j < NPG; ++j)
    s += fmaxf(nm[j] * ag[(size_t)j * HF + c] + bc, 0.0f);
  out[(size_t)g * 1024 + colbase + c]       = s * (1.0f / (float)NPG);
  out[(size_t)g * 1024 + colbase + HF + c]  = rootH[(size_t)g * HF + c];
}

// ---------------------------------------------------------------------------
extern "C" void kernel_launch(void* const* d_in, const int* in_sizes, int n_in,
                              void* d_out, int out_size, void* d_ws, size_t ws_size,
                              hipStream_t stream) {
  const float* X = (const float*)d_in[0];
  const float* w1[2] = {(const float*)d_in[1], (const float*)d_in[5]};
  const float* b1[2] = {(const float*)d_in[2], (const float*)d_in[6]};
  const float* w2[2] = {(const float*)d_in[3], (const float*)d_in[7]};
  const float* b2[2] = {(const float*)d_in[4], (const float*)d_in[8]};
  const int* srcs[2] = {(const int*)d_in[9],  (const int*)d_in[11]};
  const int* dsts[2] = {(const int*)d_in[10], (const int*)d_in[12]};

  const int N = in_sizes[0] / IN_F;      // 32768 nodes
  const int E = in_sizes[9];             // 65280 edges (tree + self-loops)
  const int G = N / NPG;                 // 256 graphs
  float* out = (float*)d_out;

  // ---- workspace carve-up (256B aligned) ----
  size_t off = 0;
  auto carve = [&](size_t bytes) -> char* {
    char* p = (char*)d_ws + off;
    off += (bytes + 255) & ~(size_t)255;
    return p;
  };
  _Float16* Xh     = (_Float16*)carve((size_t)N * IN_F * 2);   // 50.3 MB
  _Float16* W1p[2] = {(_Float16*)carve((size_t)IN_F * HF * 2),
                      (_Float16*)carve((size_t)IN_F * HF * 2)};
  _Float16* W2p[2] = {(_Float16*)carve((size_t)HF * HF * 2),
                      (_Float16*)carve((size_t)HF * HF * 2)};
  float* nrm[2] = {(float*)carve((size_t)N * 4), (float*)carve((size_t)N * 4)};
  float* U      = (float*)carve((size_t)N * HF * 4);           // 33.5 MB
  float* agg    = (float*)carve((size_t)N * HF * 4);           // 33.5 MB
  _Float16* h16 = (_Float16*)carve((size_t)N * HF * 2);        // 16.8 MB
  float* rG     = (float*)carve((size_t)G * HF * 4);
  float* rootH  = (float*)carve((size_t)G * HF * 4);

  // ---- degrees -> norms (both branches) ----
  hipMemsetAsync(nrm[0], 0, (size_t)N * 4, stream);
  hipMemsetAsync(nrm[1], 0, (size_t)N * 4, stream);
  bigcn_deg_kernel<<<(E + 255) / 256, 256, 0, stream>>>(dsts[0], E, nrm[0]);
  bigcn_deg_kernel<<<(E + 255) / 256, 256, 0, stream>>>(dsts[1], E, nrm[1]);
  bigcn_norm_kernel<<<(N + 255) / 256, 256, 0, stream>>>(nrm[0], N);
  bigcn_norm_kernel<<<(N + 255) / 256, 256, 0, stream>>>(nrm[1], N);

  // ---- X -> f16 once; pack all weights once ----
  {
    int n4 = (N * IN_F) / 4;
    bigcn_cvt_f16_kernel<<<(n4 + 255) / 256, 256, 0, stream>>>(X, Xh, n4);
  }
  for (int b = 0; b < 2; ++b) {
    bigcn_pack_w_kernel<<<(IN_F * HF + 255) / 256, 256, 0, stream>>>(
        w1[b], IN_F, 0, W1p[b]);
    bigcn_pack_w_kernel<<<(HF * HF + 255) / 256, 256, 0, stream>>>(
        w2[b], HF, 0, W2p[b]);   // W2a = rows [0, 256)
  }

  // ---- two branches: TD (colbase 0), BU (colbase 512) ----
  for (int b = 0; b < 2; ++b) {
    // Layer 1: U = X @ W1 (WMMA), scatter with norm, relu epilogue.
    bigcn_gemm_wmma_kernel<<<N / 32, 128, 0, stream>>>(Xh, IN_F, W1p[b], U);
    hipMemsetAsync(agg, 0, (size_t)N * HF * 4, stream);
    bigcn_scatter_kernel<<<E, HF, 0, stream>>>(srcs[b], dsts[b], U, nrm[b],
                                               nullptr, agg);
    bigcn_finish_h_kernel<<<N, HF, 0, stream>>>(agg, nrm[b], b1[b], h16, rootH);

    // Root projection through W2 rows [256, 1024): exact f32.
    bigcn_rootproj_kernel<<<G, HF, 0, stream>>>(X, w2[b], rG);

    // Layer 2: Q = h @ W2a (WMMA), scatter norm*(Q + rG), fused readout.
    bigcn_gemm_wmma_kernel<<<N / 32, 128, 0, stream>>>(h16, HF, W2p[b], U);
    hipMemsetAsync(agg, 0, (size_t)N * HF * 4, stream);
    bigcn_scatter_kernel<<<E, HF, 0, stream>>>(srcs[b], dsts[b], U, nrm[b],
                                               rG, agg);
    bigcn_readout_kernel<<<G, HF, 0, stream>>>(agg, nrm[b], b2[b], rootH, out,
                                               b == 0 ? 0 : 512);
  }
}
